// STBilinearMessagingPassing_26594437496909
// MI455X (gfx1250) — matrix operand order repaired and checked by reading, not compiled, not run on hardware
//
#include <hip/hip_runtime.h>
#include <hip/hip_bf16.h>

// STBilinearMessagePassing for MI455X (gfx1250, wave32, WMMA).
// B=8, TN=3072, D=1024, M=512. All GEMMs via v_wmma_f32_16x16x32_bf16.
// Flash-style online softmax: S (8x3072x3072) is never materialized.
// Epilogues transpose C-tiles through LDS so all global stores are b128.
// Streamed B-operands get WGP-scope global_prefetch_b8 (L2 -> WGP$).

typedef __bf16 bf16_t;
typedef __attribute__((ext_vector_type(16))) __bf16 v16bf;
typedef __attribute__((ext_vector_type(8)))  float  v8f;

#define B_   8
#define TN_  3072
#define D_   1024
#define M_   512

// ---- WMMA helpers -----------------------------------------------------------

__device__ __forceinline__ v8f wmma_bf16(v16bf a, v16bf b, v8f c) {
  // D = A(16x32 bf16) x B(32x16 bf16) + C(16x16 f32)
  return __builtin_amdgcn_wmma_f32_16x16x32_bf16(
      /*neg_a=*/false, a, /*neg_b=*/false, b,
      /*c_mod=*/(short)0, c, /*reuse_a=*/false, /*reuse_b=*/false);
}

// ISA 16-bit A-matrix 16x32 layout: element e of lane l holds
// K = (e>>3)*16 + (l>>4)*8 + (e&7), M = l&15.  `row` points at [m][k0].
__device__ __forceinline__ v16bf load_a(const bf16_t* row, int hi) {
  v16bf a;
#pragma unroll
  for (int e = 0; e < 16; ++e)
    a[e] = row[((e >> 3) << 4) + hi * 8 + (e & 7)];
  return a;
}

// B-operand (32x16): lane = N (l&15), element e holds K = (e>>3)*16 + (l>>4)*8 + (e&7).
__device__ __forceinline__ v16bf load_b(const bf16_t* p, int ld, int hi) {
  v16bf b;
#pragma unroll
  for (int e = 0; e < 16; ++e)
    b[e] = p[(size_t)(((e >> 3) << 4) + hi * 8 + (e & 7)) * ld];
  return b;
}

// ---- kernel 0: weights -> bf16, K-major transposes --------------------------

__global__ __launch_bounds__(256) void prep_weights(
    const float* __restrict__ U_w, const float* __restrict__ V_w,
    const float* __restrict__ W_e2, const float* __restrict__ W_e1,
    bf16_t* __restrict__ UwT, bf16_t* __restrict__ VwT,
    bf16_t* __restrict__ We2T, bf16_t* __restrict__ We1T) {
  int job = blockIdx.y;
  int i = blockIdx.x * 256 + threadIdx.x;
  if (job == 0) {                         // UwT[k][d] = U_w[d][k]
    if (i < D_ * D_) { int k = i / D_, d = i % D_; UwT[i] = (bf16_t)U_w[d * D_ + k]; }
  } else if (job == 1) {                  // VwT[k][d] = V_w[d][k]
    if (i < D_ * D_) { int k = i / D_, d = i % D_; VwT[i] = (bf16_t)V_w[d * D_ + k]; }
  } else if (job == 2) {                  // We2T[k][m] = W_e2[m][k]  (W_e2: [M][D])
    if (i < D_ * M_) { int k = i / M_, m = i % M_; We2T[i] = (bf16_t)W_e2[m * D_ + k]; }
  } else {                                // We1T[m][d] = W_e1[d][m]  (W_e1: [D][M])
    if (i < M_ * D_) { int m = i / D_, d = i % D_; We1T[i] = (bf16_t)W_e1[d * M_ + m]; }
  }
}

// ---- kernel 1: projections fU, fVT (transposed), fW -------------------------
// grid.x = (B*TN)/16 row tiles, grid.y = mode (0:U 1:V 2:W), 128 threads.

__global__ __launch_bounds__(128) void proj_kernel(
    const float* __restrict__ feat,
    const float* __restrict__ U_b, const float* __restrict__ V_b,
    const float* __restrict__ w_a,
    const bf16_t* __restrict__ UwT, const bf16_t* __restrict__ VwT,
    const bf16_t* __restrict__ We2T,
    bf16_t* __restrict__ fU, bf16_t* __restrict__ fVT, bf16_t* __restrict__ fW) {
  const int tid = threadIdx.x;
  const int lane = tid & 31, wv = tid >> 5;
  const int hi = lane >> 4, ln = lane & 15;
  const int mode = blockIdx.y;
  const int rg0 = blockIdx.x * 16;        // global row (flat over B*TN)

  __shared__ __align__(16) bf16_t fl[16 * D_];   // 32 KB feature tile (bf16)
  __shared__ __align__(16) bf16_t stg[4][256];   // per-wave 16x16 store-transpose
  for (int i = tid; i < 16 * D_; i += 128)
    fl[i] = (bf16_t)feat[(size_t)rg0 * D_ + i];
  __syncthreads();

  const int Wout = (mode == 2) ? M_ : D_;
  const bf16_t* WT = (mode == 0) ? UwT : (mode == 1) ? VwT : We2T;

  for (int ct = wv; ct < Wout / 16; ct += 4) {
    const int c0 = ct * 16;
    v8f acc = {0.f, 0.f, 0.f, 0.f, 0.f, 0.f, 0.f, 0.f};
    for (int k0 = 0; k0 < D_; k0 += 32) {
      if (k0 + 32 < D_)
        __builtin_prefetch(&WT[(size_t)(k0 + 32 + lane) * Wout + c0], 0, 3);
      v16bf a = load_a(&fl[ln * D_ + k0], hi);
      v16bf b = load_b(&WT[(size_t)k0 * Wout + c0 + ln], Wout, hi);
      acc = wmma_bf16(a, b, acc);
    }
    const int col = c0 + ln;
    if (mode == 1) {
      // fVT: [B][D][TN]; each lane owns 8 consecutive tn -> one b128 store
      const float bi = V_b[col];
      const int bidx = rg0 / TN_, tn = rg0 % TN_;
      union { uint4 q; bf16_t h[8]; } pk;
#pragma unroll
      for (int r = 0; r < 8; ++r) pk.h[r] = (bf16_t)(acc[r] + bi);
      *reinterpret_cast<uint4*>(
          &fVT[((size_t)bidx * D_ + col) * TN_ + tn + 8 * hi]) = pk.q;
    } else {
      // transpose C-tile through LDS (same wave, in-order DS) -> b128 stores
      if (mode == 0) {
        const float bi = U_b[col], wa = w_a[col];
#pragma unroll
        for (int r = 0; r < 8; ++r)
          stg[wv][(r + 8 * hi) * 16 + ln] = (bf16_t)((acc[r] + bi) * wa);
      } else {
#pragma unroll
        for (int r = 0; r < 8; ++r)
          stg[wv][(r + 8 * hi) * 16 + ln] = (bf16_t)acc[r];
      }
      bf16_t* G = (mode == 0) ? fU : fW;
      const int srow = lane >> 1, shc = (lane & 1) * 8;
      uint4 q = *reinterpret_cast<uint4*>(&stg[wv][srow * 16 + shc]);
      *reinterpret_cast<uint4*>(
          &G[(size_t)(rg0 + srow) * Wout + c0 + shc]) = q;
    }
  }
}

// ---- kernel 2: flash attention + LN + ReLU + final GEMM ---------------------
// One 16-row query tile per workgroup; 4 waves; Bc = 64 keys / iteration.
// Wave w owns j-subtile w of S and M-columns [w*128, w*128+128) of fe.

__global__ __launch_bounds__(128) void attn_kernel(
    const unsigned char* __restrict__ mask,   // jnp bool -> 1 byte/elem
    const float* __restrict__ ln_g, const float* __restrict__ ln_b,
    const bf16_t* __restrict__ fU, const bf16_t* __restrict__ fVT,
    const bf16_t* __restrict__ fW, const bf16_t* __restrict__ We1T,
    float* __restrict__ out) {
  const int tid = threadIdx.x;
  const int lane = tid & 31, wv = tid >> 5;
  const int hi = lane >> 4, ln = lane & 15;
  const int rt = blockIdx.x;
  const int bidx = rt / (TN_ / 16);
  const int rg0 = rt * 16;                    // flat row over B*TN
  const int tn0 = (rt % (TN_ / 16)) * 16;     // row within batch

  // fU tile (32KB) during main loop; reused as f32 fe tile / store staging.
  __shared__ __align__(16) char smem_raw[16 * D_ * sizeof(bf16_t)];
  bf16_t* fUt = (bf16_t*)smem_raw;
  float*  feS = (float*)smem_raw;
  __shared__ __align__(16) bf16_t Pl[16 * 64];        // P tile (C->A relayout)
  __shared__ __align__(16) bf16_t actS[16 * M_];      // relu(LN(fe)) in bf16
  __shared__ float wmaxS[4][16], wsumS[4][16], mprev[16], lsum[16];
  __shared__ float psum[128], psq[128], meanr[16], invr[16];

  {  // cooperative 128-bit copy of the fU row tile into LDS
    const uint4* src = (const uint4*)(fU + (size_t)rg0 * D_);
    uint4* dst = (uint4*)fUt;
    for (int i = tid; i < 16 * D_ * (int)sizeof(bf16_t) / 16; i += 128) dst[i] = src[i];
  }
  if (tid < 16) { mprev[tid] = -1e30f; lsum[tid] = 0.f; }
  __syncthreads();

  v8f acc[8];
#pragma unroll
  for (int t = 0; t < 8; ++t) acc[t] = (v8f){0.f, 0.f, 0.f, 0.f, 0.f, 0.f, 0.f, 0.f};

  const size_t maskbase = ((size_t)bidx * TN_ + tn0) * TN_;
  const size_t fVTbase  = (size_t)bidx * D_ * TN_;

  for (int j0 = 0; j0 < TN_; j0 += 64) {
    const int jw = j0 + wv * 16;
    // prefetch this iteration's fW rows (consumed after the softmax phase)
    __builtin_prefetch(&fW[((size_t)bidx * TN_ + j0 + lane) * M_ + wv * 128], 0, 3);
    __builtin_prefetch(&fW[((size_t)bidx * TN_ + j0 + 32 + lane) * M_ + wv * 128], 0, 3);

    // S subtile: fU(16x1024) . fV[jw:jw+16]^T, K-loop of 32 WMMAs
    v8f s = {0.f, 0.f, 0.f, 0.f, 0.f, 0.f, 0.f, 0.f};
    for (int k0 = 0; k0 < D_; k0 += 32) {
      if (k0 + 32 < D_)
        __builtin_prefetch(&fVT[fVTbase + (size_t)(k0 + 32 + lane) * TN_ + jw], 0, 3);
      v16bf a = load_a(&fUt[ln * D_ + k0], hi);
      v16bf b = load_b(&fVT[fVTbase + (size_t)k0 * TN_ + jw + ln], TN_, hi);
      s = wmma_bf16(a, b, s);
    }
    // mask + per-row max (cols live across the 16 lanes of each half-wave)
    const int jc = jw + ln;
    float rm[8];
#pragma unroll
    for (int r = 0; r < 8; ++r) {
      if (mask[maskbase + (size_t)(r + 8 * hi) * TN_ + jc]) s[r] = -1e30f;
      rm[r] = s[r];
    }
#pragma unroll
    for (int mdist = 1; mdist < 16; mdist <<= 1)
#pragma unroll
      for (int r = 0; r < 8; ++r) rm[r] = fmaxf(rm[r], __shfl_xor(rm[r], mdist, 32));
    if (ln == 0) {
#pragma unroll
      for (int r = 0; r < 8; ++r) wmaxS[wv][8 * hi + r] = rm[r];
    }
    __syncthreads();

    // combined new max; P = exp(s - m_new); rescale accumulators
    float p[8], rs[8], scl[8];
#pragma unroll
    for (int r = 0; r < 8; ++r) {
      const int row = r + 8 * hi;
      const float mo = mprev[row];
      float mn = mo;
#pragma unroll
      for (int w = 0; w < 4; ++w) mn = fmaxf(mn, wmaxS[w][row]);
      scl[r] = __expf(mo - mn);
      p[r] = __expf(s[r] - mn);
      rs[r] = p[r];
      Pl[row * 64 + wv * 16 + ln] = (bf16_t)p[r];
    }
#pragma unroll
    for (int mdist = 1; mdist < 16; mdist <<= 1)
#pragma unroll
      for (int r = 0; r < 8; ++r) rs[r] += __shfl_xor(rs[r], mdist, 32);
    if (ln == 0) {
#pragma unroll
      for (int r = 0; r < 8; ++r) wsumS[wv][8 * hi + r] = rs[r];
    }
#pragma unroll
    for (int t = 0; t < 8; ++t)
#pragma unroll
      for (int r = 0; r < 8; ++r) acc[t][r] *= scl[r];
    __syncthreads();

    if (tid < 16) {  // update running softmax stats once
      const float mo = mprev[tid];
      float mn = mo, sum = 0.f;
#pragma unroll
      for (int w = 0; w < 4; ++w) { mn = fmaxf(mn, wmaxS[w][tid]); sum += wsumS[w][tid]; }
      lsum[tid] = lsum[tid] * __expf(mo - mn) + sum;
      mprev[tid] = mn;
    }

    // fe += P(16x64) . fW[j0:j0+64][wv*128 : wv*128+128]
#pragma unroll
    for (int kk = 0; kk < 2; ++kk) {
      v16bf a = load_a(&Pl[ln * 64 + kk * 32], hi);
      const size_t frow = ((size_t)bidx * TN_ + j0 + kk * 32) * M_;
#pragma unroll
      for (int t = 0; t < 8; ++t) {
        v16bf b = load_b(&fW[frow + wv * 128 + t * 16 + ln], M_, hi);
        acc[t] = wmma_bf16(a, b, acc[t]);
      }
    }
    __syncthreads();  // Pl/wmaxS reuse + mprev/lsum visibility next iteration
  }

  // fe = acc / l  -> LDS (overlaps fUt, last read before final barrier above)
#pragma unroll
  for (int t = 0; t < 8; ++t)
#pragma unroll
    for (int r = 0; r < 8; ++r) {
      const int row = r + 8 * hi;
      feS[row * M_ + wv * 128 + t * 16 + ln] = acc[t][r] / lsum[row];
    }
  __syncthreads();

  // LayerNorm stats over M=512 (biased variance, eps=1e-5)
  {
    const int row = tid & 15, part = tid >> 4;
    float sum = 0.f, sq = 0.f;
    for (int c = part * 64; c < part * 64 + 64; ++c) {
      const float v = feS[row * M_ + c];
      sum += v; sq += v * v;
    }
    psum[tid] = sum; psq[tid] = sq;
  }
  __syncthreads();
  if (tid < 16) {
    float sum = 0.f, sq = 0.f;
#pragma unroll
    for (int p8 = 0; p8 < 8; ++p8) { sum += psum[p8 * 16 + tid]; sq += psq[p8 * 16 + tid]; }
    const float mu = sum / (float)M_;
    meanr[tid] = mu;
    invr[tid] = rsqrtf(sq / (float)M_ - mu * mu + 1e-5f);
  }
  __syncthreads();
  for (int e = tid; e < 16 * M_; e += 128) {
    const int row = e >> 9, col = e & (M_ - 1);
    const float v = (feS[e] - meanr[row]) * invr[row] * ln_g[col] + ln_b[col];
    actS[e] = (bf16_t)fmaxf(v, 0.f);
  }
  __syncthreads();   // after this, feS region is free for store staging

  // out[16 x 1024] = act(16x512) . W_e1^T  (We1T is [M][D], K-major)
  float* stg = feS + wv * 256;            // per-wave 16x16 f32 staging (1KB)
  const int srow = lane >> 1, shc = (lane & 1) * 8;
  for (int dt = wv; dt < D_ / 16; dt += 4) {
    const int d0 = dt * 16;
    v8f o = {0.f, 0.f, 0.f, 0.f, 0.f, 0.f, 0.f, 0.f};
    for (int k0 = 0; k0 < M_; k0 += 32) {
      if (k0 + 32 < M_)
        __builtin_prefetch(&We1T[(size_t)(k0 + 32 + lane) * D_ + d0], 0, 3);
      v16bf a = load_a(&actS[ln * M_ + k0], hi);
      v16bf b = load_b(&We1T[(size_t)k0 * D_ + d0 + ln], D_, hi);
      o = wmma_bf16(a, b, o);
    }
    // transpose tile through LDS (same wave, in-order DS) -> coalesced b128
#pragma unroll
    for (int r = 0; r < 8; ++r) stg[(r + 8 * hi) * 16 + ln] = o[r];
    uint4 q0 = *reinterpret_cast<uint4*>(&stg[srow * 16 + shc]);
    uint4 q1 = *reinterpret_cast<uint4*>(&stg[srow * 16 + shc + 4]);
    *reinterpret_cast<uint4*>(&out[(size_t)(rg0 + srow) * D_ + d0 + shc]) = q0;
    *reinterpret_cast<uint4*>(&out[(size_t)(rg0 + srow) * D_ + d0 + shc + 4]) = q1;
  }
}

// ---- launch -----------------------------------------------------------------

extern "C" void kernel_launch(void* const* d_in, const int* in_sizes, int n_in,
                              void* d_out, int out_size, void* d_ws, size_t ws_size,
                              hipStream_t stream) {
  const float* feature = (const float*)d_in[0];
  const unsigned char* mask = (const unsigned char*)d_in[1];
  const float* U_w = (const float*)d_in[2];
  const float* U_b = (const float*)d_in[3];
  const float* V_w = (const float*)d_in[4];
  const float* V_b = (const float*)d_in[5];
  const float* w_a = (const float*)d_in[6];
  const float* W_e2 = (const float*)d_in[7];
  const float* W_e1 = (const float*)d_in[8];
  const float* ln_g = (const float*)d_in[9];
  const float* ln_b = (const float*)d_in[10];

  char* ws = (char*)d_ws;
  const size_t nRows = (size_t)B_ * TN_;               // 24576
  bf16_t* fU   = (bf16_t*)(ws);                        // 50.3 MB
  bf16_t* fVT  = (bf16_t*)(ws + nRows * D_ * 2);       // 50.3 MB
  bf16_t* fW   = (bf16_t*)(ws + nRows * D_ * 4);       // 25.2 MB
  char* wbase  = ws + nRows * D_ * 4 + nRows * M_ * 2;
  bf16_t* UwT  = (bf16_t*)(wbase);
  bf16_t* VwT  = (bf16_t*)(wbase + (size_t)D_ * D_ * 2);
  bf16_t* We2T = (bf16_t*)(wbase + (size_t)D_ * D_ * 4);
  bf16_t* We1T = (bf16_t*)(wbase + (size_t)D_ * D_ * 4 + (size_t)D_ * M_ * 2);

  prep_weights<<<dim3((D_ * D_ + 255) / 256, 4), 256, 0, stream>>>(
      U_w, V_w, W_e2, W_e1, UwT, VwT, We2T, We1T);
  proj_kernel<<<dim3((int)(nRows / 16), 3), 128, 0, stream>>>(
      feature, U_b, V_b, w_a, UwT, VwT, We2T, fU, fVT, fW);
  attn_kernel<<<dim3((int)(nRows / 16)), 128, 0, stream>>>(
      mask, ln_g, ln_b, fU, fVT, fW, We1T, (float*)d_out);
}
